// Doc_Attention_80753975099925
// MI455X (gfx1250) — matrix operand order repaired and checked by reading, not compiled, not run on hardware
//
#include <hip/hip_runtime.h>
#include <math.h>

// Problem dims (fixed by the reference)
#define NB  32
#define NTK 8192
#define NH  256
#define NH2 512

// ---- CDNA5 WMMA types ----
typedef __attribute__((ext_vector_type(16))) __bf16 v16bf;
typedef __attribute__((ext_vector_type(8)))  float  v8f;

struct __align__(16) bf16x8 { __bf16 h[8]; };
struct __align__(8)  bf16x4 { __bf16 h[4]; };

// Branchless tanh: 1 - 2/(exp(2x)+1).  Saturates to +/-1 for large |x|,
// compiles to v_exp_f32 + v_rcp_f32 + 3 VALU ops (no EXEC manipulation).
__device__ inline float fast_tanhf(float x) {
    float e = __expf(2.0f * x);
    return 1.0f - 2.0f * __builtin_amdgcn_rcpf(e + 1.0f);
}

// =====================================================================
// Kernel 1: dec_fea[b,g] = s_t_hat[b,:] @ Wd[g,:] + bd[g]   (tiny)
// =====================================================================
__global__ void dec_proj_kernel(const float* __restrict__ s_t_hat,
                                const float* __restrict__ Wd,
                                const float* __restrict__ bd,
                                float* __restrict__ dec_fea) {
    const int b = blockIdx.x;
    const int g = threadIdx.x;
    __shared__ float ss[NH2];
    ss[g]      = s_t_hat[b * NH2 + g];
    ss[g + NH] = s_t_hat[b * NH2 + NH + g];
    __syncthreads();
    float acc = bd[g];
    const float4* wp = reinterpret_cast<const float4*>(Wd + (size_t)g * NH2);
    #pragma unroll 4
    for (int k = 0; k < NH2 / 4; ++k) {
        float4 w = wp[k];
        acc += ss[4*k+0]*w.x + ss[4*k+1]*w.y + ss[4*k+2]*w.z + ss[4*k+3]*w.w;
    }
    dec_fea[b * NH + g] = acc;
}

// =====================================================================
// Kernel 2: scores[b,t] = sum_g v[g]*tanh( (E@We^T)[b,t,g] + dec_fea[b,g]
//                                          + cov[b,t]*wc[g] )
// bf16 WMMA GEMM (f32 accumulate) fused with the tanh/dot-v epilogue.
// One wave = 16 t-rows x 256 g, 2 tiles per wave. 8 waves / block.
// We staged once per block in 128KB LDS (bf16).
// =====================================================================
__global__ void scores_kernel(const float* __restrict__ E,
                              const float* __restrict__ We,
                              const float* __restrict__ coverage,
                              const float* __restrict__ dec_fea,
                              const float* __restrict__ wc,
                              const float* __restrict__ vvec,
                              float* __restrict__ scores) {
    extern __shared__ char smraw[];
    __bf16* smWe  = reinterpret_cast<__bf16*>(smraw);           // 256*256 bf16 = 128KB
    float*  smDec = reinterpret_cast<float*>(smraw + NH * NH * 2);
    float*  smWc  = smDec + NH;
    float*  smV   = smWc + NH;

    const int b   = blockIdx.y;
    const int tid = threadIdx.x;

    // Cooperative fill: We -> bf16 in LDS (row-major [g][h]), vectorized x4.
    {
        const float4* We4 = reinterpret_cast<const float4*>(We);
        bf16x4* sm4 = reinterpret_cast<bf16x4*>(smWe);
        for (int i = tid; i < NH * NH / 4; i += 256) {
            float4 w = We4[i];
            bf16x4 o;
            o.h[0] = (__bf16)w.x; o.h[1] = (__bf16)w.y;
            o.h[2] = (__bf16)w.z; o.h[3] = (__bf16)w.w;
            sm4[i] = o;
        }
    }
    if (tid < NH) {
        smDec[tid] = dec_fea[b * NH + tid];
        smWc[tid]  = wc[tid];
        smV[tid]   = vvec[tid];
    }
    __syncthreads();

    const int wave = tid >> 5;
    const int lane = tid & 31;
    const int half = lane >> 4;   // K-half selector
    const int l15  = lane & 15;

    for (int tile = 0; tile < 2; ++tile) {
        const int t0 = ((blockIdx.x * 2 + tile) * 8 + wave) * 16;

        // ---- A fragments: 16 rows (t0..t0+15) x 256 cols of E, fp32->bf16 ----
        // ISA 16-bit A layout: lanes0-15 j<8 -> K=j, j>=8 -> K=j+8; lanes16-31 +8.
        const float* Erow = E + ((size_t)b * NTK + (size_t)(t0 + l15)) * NH;
        v16bf A[8];
        #pragma unroll
        for (int kf = 0; kf < 8; ++kf) {
            const int h0 = kf * 32 + half * 8;
            const float4* p0 = reinterpret_cast<const float4*>(Erow + h0);
            const float4* p1 = reinterpret_cast<const float4*>(Erow + h0 + 16);
            float4 c0 = p0[0], c1 = p0[1], c2 = p1[0], c3 = p1[1];
            A[kf][0]  = (__bf16)c0.x; A[kf][1]  = (__bf16)c0.y;
            A[kf][2]  = (__bf16)c0.z; A[kf][3]  = (__bf16)c0.w;
            A[kf][4]  = (__bf16)c1.x; A[kf][5]  = (__bf16)c1.y;
            A[kf][6]  = (__bf16)c1.z; A[kf][7]  = (__bf16)c1.w;
            A[kf][8]  = (__bf16)c2.x; A[kf][9]  = (__bf16)c2.y;
            A[kf][10] = (__bf16)c2.z; A[kf][11] = (__bf16)c2.w;
            A[kf][12] = (__bf16)c3.x; A[kf][13] = (__bf16)c3.y;
            A[kf][14] = (__bf16)c3.z; A[kf][15] = (__bf16)c3.w;
        }

        // coverage for the 8 M-rows this lane's half owns (M = r + 8*half)
        float cov[8];
        #pragma unroll
        for (int r = 0; r < 8; ++r)
            cov[r] = coverage[(size_t)b * NTK + t0 + r + half * 8];

        float part[8] = {0.f, 0.f, 0.f, 0.f, 0.f, 0.f, 0.f, 0.f};

        for (int nt = 0; nt < 16; ++nt) {
            const int g = nt * 16 + l15;  // N index for this lane
            v8f acc = {0.f, 0.f, 0.f, 0.f, 0.f, 0.f, 0.f, 0.f};
            #pragma unroll
            for (int kf = 0; kf < 8; ++kf) {
                // B layout: lane gives N=l15; j<8 -> K = j + 8*half, j>=8 -> K = j + 8 + 8*half.
                const int h0 = kf * 32 + half * 8;
                bf16x8 b0 = *reinterpret_cast<const bf16x8*>(smWe + g * NH + h0);
                bf16x8 b1 = *reinterpret_cast<const bf16x8*>(smWe + g * NH + h0 + 16);
                v16bf Bf;
                #pragma unroll
                for (int j = 0; j < 8; ++j) { Bf[j] = b0.h[j]; Bf[j + 8] = b1.h[j]; }
                acc = __builtin_amdgcn_wmma_f32_16x16x32_bf16(
                    false, A[kf], false, Bf, (short)0, acc, false, false);
            }
            // Fused epilogue: D layout -> VGPR r holds (M=r+8*half, N=g)
            const float dfg = smDec[g];
            const float wcg = smWc[g];
            const float vg  = smV[g];
            #pragma unroll
            for (int r = 0; r < 8; ++r) {
                float x = acc[r] + dfg + cov[r] * wcg;
                part[r] += vg * fast_tanhf(x);
            }
        }

        // Reduce over N across the 16 lanes of each half (xor masks stay in-half)
        #pragma unroll
        for (int r = 0; r < 8; ++r) {
            float p = part[r];
            p += __shfl_xor(p, 1, 32);
            p += __shfl_xor(p, 2, 32);
            p += __shfl_xor(p, 4, 32);
            p += __shfl_xor(p, 8, 32);
            part[r] = p;
        }
        if (l15 == 0) {
            #pragma unroll
            for (int r = 0; r < 8; ++r)
                scores[(size_t)b * NTK + t0 + half * 8 + r] = part[r];
        }
    }
}

// =====================================================================
// Kernel 3: in-place softmax + mask + renorm over TK, plus new_cov
// =====================================================================
__device__ inline float waveMax(float v) {
    #pragma unroll
    for (int m = 16; m >= 1; m >>= 1) v = fmaxf(v, __shfl_xor(v, m, 32));
    return v;
}
__device__ inline float waveSum(float v) {
    #pragma unroll
    for (int m = 16; m >= 1; m >>= 1) v += __shfl_xor(v, m, 32);
    return v;
}

__global__ void softmax_kernel(const float* __restrict__ mask,
                               const float* __restrict__ coverage,
                               float* __restrict__ attn,      // in: scores, out: attn
                               float* __restrict__ new_cov) {
    const int b    = blockIdx.x;
    const int tid  = threadIdx.x;
    const int wid  = tid >> 5;
    const int lane = tid & 31;
    const size_t base = (size_t)b * NTK;
    __shared__ float red[8];

    float x[32];
    #pragma unroll
    for (int i = 0; i < 32; ++i) x[i] = attn[base + tid + i * 256];

    float m = x[0];
    #pragma unroll
    for (int i = 1; i < 32; ++i) m = fmaxf(m, x[i]);
    m = waveMax(m);
    if (lane == 0) red[wid] = m;
    __syncthreads();
    float bm = red[0];
    #pragma unroll
    for (int i = 1; i < 8; ++i) bm = fmaxf(bm, red[i]);
    __syncthreads();

    float s = 0.f;
    #pragma unroll
    for (int i = 0; i < 32; ++i) {
        float e = __expf(x[i] - bm);
        float w = e * mask[base + tid + i * 256];
        x[i] = w;
        s += w;
    }
    s = waveSum(s);
    if (lane == 0) red[wid] = s;
    __syncthreads();
    float bs = 0.f;
    #pragma unroll
    for (int i = 0; i < 8; ++i) bs += red[i];
    const float inv = 1.0f / bs;

    #pragma unroll
    for (int i = 0; i < 32; ++i) {
        const size_t idx = base + tid + i * 256;
        float a = x[i] * inv;
        attn[idx] = a;
        float nc = coverage[idx] + a;
        new_cov[idx] = fminf(fmaxf(nc, 0.f), 1.f);
    }
}

// =====================================================================
// Kernel 4/5: c_t[b,h] = sum_t attn[b,t] * E[b,t,h]  (deterministic split)
// =====================================================================
__global__ void ctx_partial_kernel(const float* __restrict__ E,
                                   const float* __restrict__ attn,
                                   float* __restrict__ partial) {
    const int b = blockIdx.y, chunk = blockIdx.x, h = threadIdx.x;
    const size_t tbase = (size_t)b * NTK + (size_t)chunk * 256;
    __shared__ float sa[256];
    sa[h] = attn[tbase + h];
    __syncthreads();
    float acc = 0.f;
    for (int t = 0; t < 256; ++t) {
        // prefetch a few rows ahead (global_prefetch_b8)
        if ((t & 7) == 0 && t + 8 < 256)
            __builtin_prefetch(&E[(tbase + t + 8) * NH + h], 0, 0);
        acc += sa[t] * E[(tbase + t) * NH + h];
    }
    partial[((size_t)b * 32 + chunk) * NH + h] = acc;
}

__global__ void ctx_reduce_kernel(const float* __restrict__ partial,
                                  float* __restrict__ c_t) {
    const int b = blockIdx.x, h = threadIdx.x;
    float acc = 0.f;
    #pragma unroll
    for (int c = 0; c < 32; ++c)
        acc += partial[((size_t)b * 32 + c) * NH + h];
    c_t[b * NH + h] = acc;
}

// =====================================================================
extern "C" void kernel_launch(void* const* d_in, const int* in_sizes, int n_in,
                              void* d_out, int out_size, void* d_ws, size_t ws_size,
                              hipStream_t stream) {
    const float* s_t_hat = (const float*)d_in[0];
    const float* E       = (const float*)d_in[1];  // [B,TK,H]
    const float* mask    = (const float*)d_in[2];
    const float* cov     = (const float*)d_in[3];
    // d_in[4] attn_dist_node_to_token: unused by the reference
    const float* We      = (const float*)d_in[5];
    const float* Wd      = (const float*)d_in[6];
    const float* bd      = (const float*)d_in[7];
    const float* wc      = (const float*)d_in[8];
    const float* v       = (const float*)d_in[9];

    float* c_t     = (float*)d_out;        // [B,H]
    float* attn    = c_t + NB * NH;        // [B,TK] (scores written here first)
    float* new_cov = attn + NB * NTK;      // [B,TK]

    float* dec_fea = (float*)d_ws;         // [B,H]
    float* partial = dec_fea + NB * NH;    // [B,32,H]

    dec_proj_kernel<<<NB, NH, 0, stream>>>(s_t_hat, Wd, bd, dec_fea);

    const int smem = NH * NH * 2 + 3 * NH * 4;  // 128KB We(bf16) + dec/wc/v
    hipFuncSetAttribute(reinterpret_cast<const void*>(scores_kernel),
                        hipFuncAttributeMaxDynamicSharedMemorySize, smem);
    // grid.x = TK / (8 waves * 16 rows * 2 tiles) = 32
    scores_kernel<<<dim3(NTK / 256, NB), 256, smem, stream>>>(
        E, We, cov, dec_fea, wc, v, attn);

    softmax_kernel<<<NB, 256, 0, stream>>>(mask, cov, attn, new_cov);

    ctx_partial_kernel<<<dim3(32, NB), NH, 0, stream>>>(E, attn, partial);
    ctx_reduce_kernel<<<NB, NH, 0, stream>>>(partial, c_t);
}
